// DistributedAFNO2D_42528766165095
// MI455X (gfx1250) — matrix operand order repaired and checked by reading, not compiled, not run on hardware
//
#include <hip/hip_runtime.h>

// ---------------- problem constants ----------------
#define HN     256          // H
#define WN     256          // W
#define WF     129          // W/2+1 rfft bins
#define WFP    132          // padded freq stride (complex elements)
#define CN     768          // hidden
#define NBLK   8
#define BLK    96
#define LAM    0.01f

typedef __attribute__((ext_vector_type(16))) __bf16 v16bf;
typedef __attribute__((ext_vector_type(8)))  float  v8f;

// gfx1250 async global->LDS path (guarded; falls back to plain loads)
#if defined(__gfx1250__) && __has_builtin(__builtin_amdgcn_global_load_async_to_lds_b64)
#define HAVE_ASYNC_LDS 1
typedef int b64vec __attribute__((vector_size(8)));                  // int __vector(2)
typedef __attribute__((address_space(1))) b64vec* gas_b64;           // global ptr
typedef __attribute__((address_space(3))) b64vec* las_b64;           // LDS ptr
#else
#define HAVE_ASYNC_LDS 0
#endif

// =====================================================================
// 256-point radix-2 complex FFT in LDS (interleaved float2).
// 64 threads per FFT (t = 0..63), 4 FFTs per 256-thread workgroup.
// tw: 128-entry table tw[j] = (cos(-2*pi*j/256), sin(-2*pi*j/256)).
// sgn = -1 forward, +1 inverse (conjugated twiddles).
// =====================================================================
__device__ __forceinline__ void fft256(float2* d, int t, float sgn, const float2* tw)
{
    // bit-reversal permutation (each swap pair executed by exactly one thread)
#pragma unroll
    for (int q = 0; q < 4; ++q) {
        int j = t + 64*q;
        int r = (int)(__brev((unsigned)j) >> 24);
        if (r > j) { float2 a = d[j]; d[j] = d[r]; d[r] = a; }
    }
    __syncthreads();
#pragma unroll
    for (int st = 1; st <= 8; ++st) {
        const int lh    = st - 1;           // log2(half)
        const int half  = 1 << lh;
        const int shift = 8 - st;           // twiddle index multiplier
#pragma unroll
        for (int q = 0; q < 2; ++q) {
            int bf  = t + 64*q;             // 128 butterflies per stage
            int grp = bf >> lh;
            int pos = bf & (half - 1);
            int i1  = (grp << st) + pos;
            int i2  = i1 + half;
            float2 w = tw[pos << shift];
            float c = w.x;
            float s = -sgn * w.y;           // fwd: w.y, inv: -w.y
            float2 v2 = d[i2];
            float tr = v2.x*c - v2.y*s;
            float ti = v2.x*s + v2.y*c;
            float2 v1 = d[i1];
            d[i1] = make_float2(v1.x + tr, v1.y + ti);
            d[i2] = make_float2(v1.x - tr, v1.y - ti);
        }
        __syncthreads();
    }
}

__device__ __forceinline__ void build_twiddles(float2* tw, int tid)
{
    if (tid < 128) {
        float s, c;
        __sincosf(-6.28318530717958647f * (float)tid * (1.0f/256.0f), &s, &c);
        tw[tid] = make_float2(c, s);
    }
}

// =====================================================================
// Pass 1: row rFFT: x (B,C,H,W) f32 -> freq (B,C,H,WFP) complex f32
// =====================================================================
__global__ void __launch_bounds__(256)
afno_rfft_rows(const float* __restrict__ x, float* __restrict__ freq)
{
    __shared__ float2 sd[4][256];
    __shared__ float2 twd[128];
    int tid = threadIdx.x;
    int sub = tid >> 6, t = tid & 63;
    build_twiddles(twd, tid);
    size_t row = (size_t)blockIdx.x * 4 + sub;     // (b*C+c)*H + h
    const float* src = x + row * (size_t)WN;
    float2* d = sd[sub];
#pragma unroll
    for (int q = 0; q < 4; ++q) { int j = t + 64*q; d[j] = make_float2(src[j], 0.0f); }
    __syncthreads();
    fft256(d, t, -1.0f, twd);
    float2* dst = (float2*)freq + row * (size_t)WFP;
    const float sc = 1.0f / 16.0f;                 // ortho: 1/sqrt(256)
    for (int w = t; w < WF; w += 64)
        dst[w] = make_float2(d[w].x * sc, d[w].y * sc);
}

// =====================================================================
// Pass 2/4: complex FFT along H for each (b,c,wf) column, in place.
// Column gather goes global -> LDS via async copy when available.
// =====================================================================
__global__ void __launch_bounds__(256)
afno_fft_cols(float* __restrict__ freq, float sgn)
{
    __shared__ float2 sd[4][256];
    __shared__ float2 twd[128];
    int tid = threadIdx.x;
    int sub = tid >> 6, t = tid & 63;
    build_twiddles(twd, tid);
    size_t col = (size_t)blockIdx.x * 4 + sub;     // (b*C+c)*WF + wf
    size_t bc  = col / WF;
    int    wf  = (int)(col - bc * WF);
    float2* base = (float2*)freq + bc * (size_t)HN * WFP + wf;
    float2* d = sd[sub];
#if HAVE_ASYNC_LDS
#pragma unroll
    for (int q = 0; q < 4; ++q) {
        int h = t + 64*q;
        __builtin_amdgcn_global_load_async_to_lds_b64(
            (gas_b64)(base + (size_t)h * WFP), (las_b64)(d + h), 0, 0);
    }
    asm volatile("s_wait_asynccnt 0" ::: "memory");
#else
#pragma unroll
    for (int q = 0; q < 4; ++q) { int h = t + 64*q; d[h] = base[(size_t)h * WFP]; }
#endif
    __syncthreads();
    fft256(d, t, sgn, twd);
    const float sc = 1.0f / 16.0f;
#pragma unroll
    for (int q = 0; q < 4; ++q) {
        int h = t + 64*q;
        base[(size_t)h * WFP] = make_float2(d[h].x * sc, d[h].y * sc);
    }
}

// =====================================================================
// Pass 3: block-diagonal complex 2-layer MLP in frequency domain (bf16 WMMA).
// Grid: 2064 WGs = (b,k) x 129; 128 threads = 4 waves; each wave does 4 tiles
// of 16 positions. In place on freq.
//
// LDS layout (dynamic, 120 KB):
//   wfrag : [layer][comp][9216] bf16  B-fragments of w1/w2 for block k
//   afrag : per wave [comp][1536] bf16 A-fragments of input tile
//   ofrag : per wave [comp][1536] bf16 A-fragments of layer-1 output
// Fragment orders follow the CDNA5 16-bit WMMA VGPR layouts so each lane
// loads its 16 contiguous bf16 (32 B) per fragment. A-fragments are hoisted
// into registers across the 6 output chunks (only weight fragments stream
// from LDS in the inner loop).
// =====================================================================
__global__ void __launch_bounds__(128)
afno_freq_mlp(float* __restrict__ freq,
              const float* __restrict__ w1, const float* __restrict__ b1,
              const float* __restrict__ w2, const float* __restrict__ b2)
{
    extern __shared__ char smem_raw[];
    __bf16* wfrag = (__bf16*)smem_raw;             // 4*9216
    __bf16* afrag = wfrag + 4*9216;                // 4 waves * 3072
    __bf16* ofrag = afrag + 4*3072;                // 4 waves * 3072

    const int tid  = threadIdx.x;
    const int lane = tid & 31;
    const int wave = tid >> 5;

    const int bk = blockIdx.x / 129;               // b*8 + k
    const int wg = blockIdx.x % 129;
    const int b  = bk >> 3;
    const int k  = bk & 7;

    // ---- stage w1/w2 (block k) into LDS as bf16 B-fragments ----
    for (int f = tid; f < 4*9216; f += 128) {
        int layer = f / 18432;
        int rem   = f % 18432;
        int comp  = rem / 9216;
        int p     = rem % 9216;
        int nc    = p / 1536;
        int q     = p % 1536;
        int kc    = q / 512;
        int l     = (q % 512) >> 4;
        int e     = q & 15;
        int K = kc*32 + (e >> 3)*16 + (l >> 4)*8 + (e & 7); // input channel 0..95
        int N = nc*16 + (l & 15);                  // output channel 0..95
        const float* w = layer ? w2 : w1;
        wfrag[f] = (__bf16)w[(((size_t)k*BLK + K)*BLK + N)*2 + comp];
    }
    __syncthreads();

    __bf16* myA = afrag + wave*3072;
    __bf16* myO = ofrag + wave*3072;

    for (int iter = 0; iter < 4; ++iter) {
        int tile = wg*16 + iter*4 + wave;          // 0..2063, unique per wave
        int n0   = tile * 16;

        // ---- stage A tile: 96 in-channels x 16 positions, f32 -> bf16 frags
        for (int f = lane; f < BLK*16; f += 32) {
            int i = f >> 4;                        // channel
            int m = f & 15;                        // position in tile
            int n = n0 + m;
            int h  = n / WF;
            int wq = n - h*WF;
            const float2* src = (const float2*)freq +
                (((size_t)b*CN + (size_t)k*BLK + i)*HN + h)*WFP + wq;
            float2 v = *src;
            // A-matrix (16x32 bf16) element placement for K=i%32, M=m
            int kc = i >> 5, ko = i & 31;
            int hi = ko >> 4, r = ko & 15;
            int lh = r >> 3;
            int e  = hi*8 + (r & 7);
            int l  = lh*16 + m;
            int idx = kc*512 + l*16 + e;
            myA[idx]        = (__bf16)v.x;
            myA[1536 + idx] = (__bf16)v.y;
        }
        asm volatile("s_wait_dscnt 0" ::: "memory");  // wave-local LDS ordering

        // hoist A fragments into registers (held across all 6 output chunks)
        v16bf a_r[3], a_i[3];
#pragma unroll
        for (int kc = 0; kc < 3; ++kc) {
            a_r[kc] = *(const v16bf*)(myA + kc*512 + lane*16);
            a_i[kc] = *(const v16bf*)(myA + 1536 + kc*512 + lane*16);
        }

        // ---------------- layer 1: o1 = relu(cmul(A, w1) + b1) ----------------
        for (int nc = 0; nc < 6; ++nc) {
            v8f rr = {}, ii = {}, ir = {}, ri = {};
#pragma unroll
            for (int kc = 0; kc < 3; ++kc) {
                const __bf16* wb = wfrag + nc*1536 + kc*512 + lane*16;
                v16bf wr = *(const v16bf*)wb;
                v16bf wi = *(const v16bf*)(wb + 9216);
                rr = __builtin_amdgcn_wmma_f32_16x16x32_bf16(false, a_r[kc], false, wr, (short)0, rr, false, false);
                ii = __builtin_amdgcn_wmma_f32_16x16x32_bf16(false, a_i[kc], false, wi, (short)0, ii, false, false);
                ir = __builtin_amdgcn_wmma_f32_16x16x32_bf16(false, a_i[kc], false, wr, (short)0, ir, false, false);
                ri = __builtin_amdgcn_wmma_f32_16x16x32_bf16(false, a_r[kc], false, wi, (short)0, ri, false, false);
            }
            int o = nc*16 + (lane & 15);
            float b1r = b1[((size_t)k*BLK + o)*2 + 0];
            float b1i = b1[((size_t)k*BLK + o)*2 + 1];
#pragma unroll
            for (int r = 0; r < 8; ++r) {
                float vr = fmaxf(rr[r] - ii[r] + b1r, 0.0f);
                float vi = fmaxf(ir[r] + ri[r] + b1i, 0.0f);
                int m = r + 8*(lane >> 4);         // D layout: M = vgpr + 8*laneHi
                // scatter into layer-2 A-fragment order (K = o, M = m)
                int kc2 = o >> 5, ko = o & 31;
                int hi2 = ko >> 4, r2 = ko & 15;
                int lh  = r2 >> 3;
                int e   = hi2*8 + (r2 & 7);
                int l   = lh*16 + m;
                int idx = kc2*512 + l*16 + e;
                myO[idx]        = (__bf16)vr;
                myO[1536 + idx] = (__bf16)vi;
            }
        }
        asm volatile("s_wait_dscnt 0" ::: "memory");

        // hoist layer-2 A fragments (o1) into registers
        v16bf o_r[3], o_i[3];
#pragma unroll
        for (int kc = 0; kc < 3; ++kc) {
            o_r[kc] = *(const v16bf*)(myO + kc*512 + lane*16);
            o_i[kc] = *(const v16bf*)(myO + 1536 + kc*512 + lane*16);
        }

        // ------- layer 2: out = softshrink(cmul(o1, w2) + b2); write back -----
        for (int nc = 0; nc < 6; ++nc) {
            v8f rr = {}, ii = {}, ir = {}, ri = {};
#pragma unroll
            for (int kc = 0; kc < 3; ++kc) {
                const __bf16* wb = wfrag + 2*9216 + nc*1536 + kc*512 + lane*16;
                v16bf wr = *(const v16bf*)wb;
                v16bf wi = *(const v16bf*)(wb + 9216);
                rr = __builtin_amdgcn_wmma_f32_16x16x32_bf16(false, o_r[kc], false, wr, (short)0, rr, false, false);
                ii = __builtin_amdgcn_wmma_f32_16x16x32_bf16(false, o_i[kc], false, wi, (short)0, ii, false, false);
                ir = __builtin_amdgcn_wmma_f32_16x16x32_bf16(false, o_i[kc], false, wr, (short)0, ir, false, false);
                ri = __builtin_amdgcn_wmma_f32_16x16x32_bf16(false, o_r[kc], false, wi, (short)0, ri, false, false);
            }
            int o = nc*16 + (lane & 15);
            float b2r = b2[((size_t)k*BLK + o)*2 + 0];
            float b2i = b2[((size_t)k*BLK + o)*2 + 1];
#pragma unroll
            for (int r = 0; r < 8; ++r) {
                float vr = rr[r] - ii[r] + b2r;
                float vi = ir[r] + ri[r] + b2i;
                vr = (vr > LAM) ? (vr - LAM) : ((vr < -LAM) ? (vr + LAM) : 0.0f);
                vi = (vi > LAM) ? (vi - LAM) : ((vi < -LAM) ? (vi + LAM) : 0.0f);
                int m = r + 8*(lane >> 4);
                int n = n0 + m;
                int h  = n / WF;
                int wq = n - h*WF;
                float2* dst = (float2*)freq +
                    (((size_t)b*CN + (size_t)k*BLK + o)*HN + h)*WFP + wq;
                *dst = make_float2(vr, vi);
            }
        }
    }
}

// =====================================================================
// Pass 5: inverse row FFT with Hermitian reconstruction + residual add.
// =====================================================================
__global__ void __launch_bounds__(256)
afno_irfft_rows(const float* __restrict__ freq, const float* __restrict__ x,
                float* __restrict__ out)
{
    __shared__ float2 sd[4][256];
    __shared__ float2 twd[128];
    int tid = threadIdx.x;
    int sub = tid >> 6, t = tid & 63;
    build_twiddles(twd, tid);
    size_t row = (size_t)blockIdx.x * 4 + sub;
    const float2* src = (const float2*)freq + row * (size_t)WFP;
    float2* d = sd[sub];
#if HAVE_ASYNC_LDS
    for (int w = t; w < WF; w += 64) {
        __builtin_amdgcn_global_load_async_to_lds_b64(
            (gas_b64)(src + w), (las_b64)(d + w), 0, 0);
    }
    asm volatile("s_wait_asynccnt 0" ::: "memory");
#else
    for (int w = t; w < WF; w += 64) d[w] = src[w];
#endif
    __syncthreads();
    for (int j = 129 + t; j < 256; j += 64)        // X[256-j] = conj(X[j])
        d[j] = make_float2(d[256 - j].x, -d[256 - j].y);
    __syncthreads();
    fft256(d, t, 1.0f, twd);
    const float sc = 1.0f / 16.0f;
    const float* bx = x   + row * (size_t)WN;
    float*       dp = out + row * (size_t)WN;
#pragma unroll
    for (int q = 0; q < 4; ++q) {
        int j = t + 64*q;
        dp[j] = d[j].x * sc + bx[j];               // real part + residual
    }
}

// =====================================================================
extern "C" void kernel_launch(void* const* d_in, const int* in_sizes, int n_in,
                              void* d_out, int out_size, void* d_ws, size_t ws_size,
                              hipStream_t stream)
{
    (void)in_sizes; (void)n_in; (void)out_size; (void)ws_size;
    const float* x  = (const float*)d_in[0];
    const float* w1 = (const float*)d_in[1];
    const float* b1 = (const float*)d_in[2];
    const float* w2 = (const float*)d_in[3];
    const float* b2 = (const float*)d_in[4];
    float* out  = (float*)d_out;
    float* freq = (float*)d_ws;   // (B*C*H*WFP) complex f32 ~ 415 MB

    const int nrows = 2 * CN * HN;   // 393216
    const int ncols = 2 * CN * WF;   // 198144

    afno_rfft_rows<<<nrows / 4, 256, 0, stream>>>(x, freq);
    afno_fft_cols <<<ncols / 4, 256, 0, stream>>>(freq, -1.0f);

    const size_t mlp_lds = (size_t)(4*9216 + 2*4*3072) * sizeof(__bf16); // 120 KB
    afno_freq_mlp<<<2 * NBLK * 129, 128, mlp_lds, stream>>>(freq, w1, b1, w2, b2);

    afno_fft_cols <<<ncols / 4, 256, 0, stream>>>(freq, 1.0f);
    afno_irfft_rows<<<nrows / 4, 256, 0, stream>>>(freq, x, out);
}
